// KolmogorovArnoldLayer_61795989455362
// MI455X (gfx1250) — matrix operand order repaired and checked by reading, not compiled, not run on hardware
//
#include <hip/hip_runtime.h>
#include <hip/hip_bf16.h>
#include <stdint.h>

// KAL: N=32768 rows, D=8, Q=17, H=32, O=256.
// Stage 1 (psi) is elementwise VALU work -> packed f16 (v_pk_fma_f16 / v_pk_max).
// Stage 2 (phi) is a GEMM M=32768, K=Q*H=544, N=256 -> v_wmma_f32_16x16x32_f16
// with fp32 accumulation. hp (A matrix) is built in registers per k-step, never
// written to memory. B (phi_w2) is pre-packed into WMMA fragment order in d_ws.
// Wave tile is 64x64 (4 M-tiles x 4 O-tiles) to balance A-build VALU vs B-load
// traffic: 16 WMMAs per k-step against only 4 B-fragment loads + 1 w1/b1 load.

typedef _Float16 v16h __attribute__((ext_vector_type(16)));
typedef float    v8f  __attribute__((ext_vector_type(8)));
typedef _Float16 h2   __attribute__((ext_vector_type(2)));

#define NROWS 32768
#define DD 8
#define QQ 17
#define HH 32
#define OO 256
#define BM 128            // rows per block
#define NBLK (NROWS/BM)   // 256 blocks

// d_ws layout (dword offsets). Total 78609 dwords ~= 314 KB (ws_size assumed >= this).
#define WS_B    0         // 17*16*2*16*8 = 69632 dwords of f16 pairs (B fragments)
#define WS_AW1  69632     // 17*32*8 = 4352 dwords (phi_w1 packed in A-frag K order)
#define WS_AB1  73984     // 4352 dwords (phi_b1 packed)
#define WS_BIAS 78336     // 256 floats: sum_q phi_b2[q][o]
#define WS_S0   78592     // 17 floats: sum_d psi_b2[q][d]

// ---------------------------------------------------------------------------
// One-time packing kernel: reformat phi weights into WMMA fragment layouts.
// B-frag (32x16 f16, wave32): same K order as A-frag: lanes 0-15 hold K {0..7,16..23},
// lanes 16-31 hold K {8..15,24..31}; N-col = lane&15.
// A-frag (16x32 f16): lane<16 -> K in {0..7,16..23}; lane>=16 -> K in {8..15,24..31};
// vgpr j pair K0 = (lane&16 ? 8:0) + (j<4 ? 2j : 8+2j).
// ---------------------------------------------------------------------------
__global__ void kal_pack(const float* __restrict__ psi_b2,
                         const float* __restrict__ phi_w1,
                         const float* __restrict__ phi_b1,
                         const float* __restrict__ phi_w2,
                         const float* __restrict__ phi_b2,
                         h2* __restrict__ wsB, h2* __restrict__ wsAw1,
                         h2* __restrict__ wsAb1, float* __restrict__ bias,
                         float* __restrict__ s0) {
  const int b = blockIdx.x, t = threadIdx.x;
  if (b < QQ * 16) {
    // B pack: block -> (q, o-tile); thread -> (half, ocol, j)
    const int q = b >> 4, ot = b & 15;
    const int half = t >> 7, ocol = (t >> 3) & 15, j = t & 7;
    const int k = half * 8 + ((j < 4) ? 2 * j : 8 + 2 * j);
    const int o = ot * 16 + ocol;
    h2 p;
    p.x = (_Float16)phi_w2[(q * HH + k) * OO + o];
    p.y = (_Float16)phi_w2[(q * HH + k + 1) * OO + o];
    wsB[(((q * 16 + ot) * 2 + half) * 16 + ocol) * 8 + j] = p;
  } else if (b < QQ * 16 + 34) {
    // A-side pack for phi_w1 / phi_b1
    const int idx = (b - QQ * 16) * 256 + t;
    if (idx < 2 * QQ * HH * 8) {
      const int arr = idx / (QQ * HH * 8);
      const int r = idx % (QQ * HH * 8);
      const int q = r >> 8, l = (r >> 3) & 31, j = r & 7;
      const int k0 = ((l & 16) ? 8 : 0) + ((j < 4) ? 2 * j : 8 + 2 * j);
      const float* src = arr ? phi_b1 : phi_w1;
      h2 p;
      p.x = (_Float16)src[q * HH + k0];
      p.y = (_Float16)src[q * HH + k0 + 1];
      (arr ? wsAb1 : wsAw1)[(q * HH + l) * 8 + j] = p;
    }
  } else {
    // bias[o] = sum_q phi_b2[q][o];  s0[q] = sum_d psi_b2[q][d]
    float acc = 0.f;
    for (int q = 0; q < QQ; ++q) acc += phi_b2[q * OO + t];
    bias[t] = acc;
    if (t < QQ) {
      float a = 0.f;
      for (int d = 0; d < DD; ++d) a += psi_b2[t * DD + d];
      s0[t] = a;
    }
  }
}

// ---------------------------------------------------------------------------
// Fused psi + phi-GEMM kernel. 256 threads (8 wave32), 128 rows per block.
// Waves arranged 2 row-groups (64 rows) x 4 col-groups (64 cols).
// ---------------------------------------------------------------------------
__global__ __launch_bounds__(256, 1) void kal_main(
    const float* __restrict__ x,
    const float* __restrict__ psi_w1, const float* __restrict__ psi_b1,
    const float* __restrict__ psi_w2,
    const h2* __restrict__ wsB, const h2* __restrict__ wsAw1,
    const h2* __restrict__ wsAb1, const float* __restrict__ bias,
    const float* __restrict__ s0,
    float* __restrict__ out) {
  __shared__ h2 lw1[QQ * 128];     // psi weights packed as (h,h+1) f16 pairs
  __shared__ h2 lb1[QQ * 128];
  __shared__ h2 lw2[QQ * 128];
  __shared__ float s_lds[BM * 18]; // s[n,q], padded stride 18

  const int t = threadIdx.x;

  // ---- stage psi weights into LDS as packed f16 pairs (broadcast-friendly) ----
  for (int i = t; i < QQ * 128; i += 256) {
    const int gb = i * 2;  // [q][d][h] flattened, pairs along h
    h2 a, bb, c;
    a.x = (_Float16)psi_w1[gb];  a.y = (_Float16)psi_w1[gb + 1];
    bb.x = (_Float16)psi_b1[gb]; bb.y = (_Float16)psi_b1[gb + 1];
    c.x = (_Float16)psi_w2[gb];  c.y = (_Float16)psi_w2[gb + 1];
    lw1[i] = a; lb1[i] = bb; lw2[i] = c;
  }

  // ---- load this thread's x row (duplicated into h2 lanes) ----
  const int r = t & 127;
  const float* xr = x + ((size_t)blockIdx.x * BM + r) * DD;
  h2 xp[DD];
#pragma unroll
  for (int d = 0; d < DD; ++d) {
    _Float16 xv = (_Float16)xr[d];
    xp[d].x = xv; xp[d].y = xv;
  }
  __syncthreads();

  // ---- psi: s[r][q] via packed f16 fma/max, f32 final reduction ----
  const int q0 = (t >> 7) ? 8 : 0;
  const int q1 = (t >> 7) ? QQ : 8;
  for (int q = q0; q < q1; ++q) {
    h2 acc2[16];
#pragma unroll
    for (int j = 0; j < 16; ++j) { acc2[j].x = (_Float16)0; acc2[j].y = (_Float16)0; }
    const h2* w1q = lw1 + q * 128;
    const h2* b1q = lb1 + q * 128;
    const h2* w2q = lw2 + q * 128;
#pragma unroll
    for (int d = 0; d < DD; ++d) {
      h2 xv = xp[d];
#pragma unroll
      for (int j = 0; j < 16; ++j) {
        h2 tv = __builtin_elementwise_fma(xv, w1q[d * 16 + j], b1q[d * 16 + j]);
        h2 z; z.x = (_Float16)0; z.y = (_Float16)0;
        tv = __builtin_elementwise_max(tv, z);                      // v_pk_max
        acc2[j] = __builtin_elementwise_fma(tv, w2q[d * 16 + j], acc2[j]);
      }
    }
    float sv = s0[q];
#pragma unroll
    for (int j = 0; j < 16; ++j) sv += (float)acc2[j].x + (float)acc2[j].y;
    s_lds[r * 18 + q] = sv;
  }
  __syncthreads();

  // ---- phi GEMM: each wave owns a 64-row x 64-col output tile ----
  const int w = t >> 5, lane = t & 31;
  const int lhalf = lane >> 4, lcol = lane & 15;
  const int rg = w >> 2;   // row group: rows [rg*64, rg*64+64)
  const int cg = w & 3;    // col group: cols [cg*64, cg*64+64)

  v8f acc[4][4];           // [m-tile][o-tile]
#pragma unroll
  for (int otl = 0; otl < 4; ++otl) {
    float bo = bias[(cg * 4 + otl) * 16 + lcol]; // bias is lane-constant per C tile
#pragma unroll
    for (int mt = 0; mt < 4; ++mt)
#pragma unroll
      for (int e = 0; e < 8; ++e) acc[mt][otl][e] = bo;
  }

  const _Float16* aw = (const _Float16*)wsAw1;
  const _Float16* ab = (const _Float16*)wsAb1;
  const _Float16* bw = (const _Float16*)wsB;

  for (int q = 0; q < QQ; ++q) {
    // phi_w1/b1 fragment constants: shared by all 4 M-tiles (depend on q, lane only)
    v16h w1v = *(const v16h*)(aw + (size_t)(q * HH + lane) * 16);
    v16h b1v = *(const v16h*)(ab + (size_t)(q * HH + lane) * 16);
    v16h z16;
#pragma unroll
    for (int e = 0; e < 16; ++e) z16[e] = (_Float16)0;

    // Build 4 A fragments in registers: hp = relu(s * w1 + b1), f16 packed.
    v16h av[4];
#pragma unroll
    for (int mt = 0; mt < 4; ++mt) {
      const float sf = s_lds[(rg * 64 + mt * 16 + lcol) * 18 + q];
      const _Float16 sh = (_Float16)sf;
      v16h sb;
#pragma unroll
      for (int e = 0; e < 16; ++e) sb[e] = sh;
      av[mt] = __builtin_elementwise_max(
          __builtin_elementwise_fma(sb, w1v, b1v), z16);
    }

#pragma unroll
    for (int otl = 0; otl < 4; ++otl) {
      const int ot = cg * 4 + otl;
      v16h bv = *(const v16h*)(bw +
          (size_t)(((q * 16 + ot) * 2 + lhalf) * 16 + lcol) * 16);
#pragma unroll
      for (int mt = 0; mt < 4; ++mt) {
        acc[mt][otl] = __builtin_amdgcn_wmma_f32_16x16x32_f16(
            false, av[mt], false, bv, (short)0, acc[mt][otl], false, false);
      }
    }
  }

  // ---- store: C layout -> lane<16: M=e, lane>=16: M=8+e; col per lane ----
#pragma unroll
  for (int mt = 0; mt < 4; ++mt) {
    const size_t grow = (size_t)blockIdx.x * BM + (size_t)rg * 64 +
                        (size_t)mt * 16 + (size_t)lhalf * 8;
#pragma unroll
    for (int otl = 0; otl < 4; ++otl) {
      const int col = (cg * 4 + otl) * 16 + lcol;
#pragma unroll
      for (int e = 0; e < 8; ++e) out[(grow + e) * OO + col] = acc[mt][otl][e];
    }
  }
}

extern "C" void kernel_launch(void* const* d_in, const int* in_sizes, int n_in,
                              void* d_out, int out_size, void* d_ws, size_t ws_size,
                              hipStream_t stream) {
  const float* x      = (const float*)d_in[0];
  const float* psi_w1 = (const float*)d_in[1];
  const float* psi_b1 = (const float*)d_in[2];
  const float* psi_w2 = (const float*)d_in[3];
  const float* psi_b2 = (const float*)d_in[4];
  const float* phi_w1 = (const float*)d_in[5];
  const float* phi_b1 = (const float*)d_in[6];
  const float* phi_w2 = (const float*)d_in[7];
  const float* phi_b2 = (const float*)d_in[8];
  (void)in_sizes; (void)n_in; (void)out_size; (void)ws_size; // needs ~314 KB of ws

  uint32_t* ws = (uint32_t*)d_ws;
  h2*    wsB   = (h2*)(ws + WS_B);
  h2*    wsAw1 = (h2*)(ws + WS_AW1);
  h2*    wsAb1 = (h2*)(ws + WS_AB1);
  float* bias  = (float*)(ws + WS_BIAS);
  float* s0    = (float*)(ws + WS_S0);

  kal_pack<<<QQ * 16 + 34 + 1, 256, 0, stream>>>(
      psi_b2, phi_w1, phi_b1, phi_w2, phi_b2, wsB, wsAw1, wsAb1, bias, s0);
  kal_main<<<NBLK, 256, 0, stream>>>(
      x, psi_w1, psi_b1, psi_w2, wsB, wsAw1, wsAb1, bias, s0, (float*)d_out);
}
